// Mamba_19267223290737
// MI455X (gfx1250) — matrix operand order repaired and checked by reading, not compiled, not run on hardware
//
#include <hip/hip_runtime.h>
#include <hip/hip_bf16.h>
#include <math.h>

// Problem dims (fixed by reference)
#define BN 2
#define SN 1024
#define HN 64
#define EN 128
#define KN 4
#define RN 4
#define LN 2
#define ROWS (BN*SN)   // 2048 token rows
#define DBC_LD 144     // 132 padded to 9*16 for WMMA N-tiling
#define SCAN_PF 8      // scan prefetch distance (timesteps)

typedef __attribute__((ext_vector_type(16))) __bf16 v16bf;
typedef __attribute__((ext_vector_type(8)))  float  v8f;

__device__ __forceinline__ float sigmoidf_(float x) { return 1.f / (1.f + __expf(-x)); }

// ---------------------------------------------------------------------------
// Weight convert + transpose to bf16:  dst[l][n][k] = src[l][k][n]  (pad n>=Nc with 0)
// ---------------------------------------------------------------------------
__global__ void wconv_kernel(const float* __restrict__ src, __bf16* __restrict__ dst,
                             int Lc, int Kd, int Nc, int Np) {
  int idx = blockIdx.x * blockDim.x + threadIdx.x;
  int total = Lc * Np * Kd;
  if (idx >= total) return;
  int k = idx % Kd;
  int n = (idx / Kd) % Np;
  int l = idx / (Kd * Np);
  float v = (n < Nc) ? src[((size_t)l * Kd + k) * Nc + n] : 0.f;
  dst[idx] = (__bf16)v;
}

__global__ void copy_kernel(const float* __restrict__ src, float* __restrict__ dst, int n) {
  int i = blockIdx.x * blockDim.x + threadIdx.x;
  if (i < n) dst[i] = src[i];
}

// ---------------------------------------------------------------------------
// RMSNorm over H=64: one wave32 per row, 2 elements per lane, shuffle reduce.
// ---------------------------------------------------------------------------
__global__ void rmsnorm_kernel(const float* __restrict__ in, const float* __restrict__ w,
                               __bf16* __restrict__ out_bf, float* __restrict__ out_f) {
  int lane = threadIdx.x & 31;
  int row  = (blockIdx.x * blockDim.x + threadIdx.x) >> 5;
  if (row >= ROWS) return;
  const float* p = in + (size_t)row * HN;
  float v0 = p[lane], v1 = p[lane + 32];
  float ss = v0 * v0 + v1 * v1;
  #pragma unroll
  for (int off = 16; off; off >>= 1) ss += __shfl_xor(ss, off, 32);
  float rs = rsqrtf(ss * (1.f / HN) + 1e-5f);
  float o0 = v0 * rs * w[lane], o1 = v1 * rs * w[lane + 32];
  if (out_bf) { out_bf[(size_t)row * HN + lane] = (__bf16)o0; out_bf[(size_t)row * HN + lane + 32] = (__bf16)o1; }
  if (out_f)  { out_f [(size_t)row * HN + lane] = o0;         out_f [(size_t)row * HN + lane + 32] = o1; }
}

// ---------------------------------------------------------------------------
// GEMM via v_wmma_f32_16x16x32_bf16. One wave computes one 16x16 tile of
// C = A (MxK, row-major bf16) x Bt^T (Bt is NxK row-major bf16) + bias (+ addend).
// Fragment layouts per CDNA5 ISA 7.12.2. Next K-block prefetched
// via global_prefetch_b8 while the current WMMA is in flight.
// ---------------------------------------------------------------------------
__global__ void gemm_wmma_bf16(const __bf16* __restrict__ A, int lda,
                               const __bf16* __restrict__ Bt, int ldb,
                               const float* __restrict__ bias,
                               const float* __restrict__ addend,
                               float* __restrict__ C, int ldc,
                               int M, int N, int Kd) {
  int lane = threadIdx.x & 31;
  int wave = (blockIdx.x * blockDim.x + threadIdx.x) >> 5;
  int tilesN = N >> 4;
  int tm = wave / tilesN, tn = wave % tilesN;
  if (tm * 16 >= M) return;                    // wave-uniform; EXEC stays all-ones otherwise

  int arow = tm * 16 + (lane & 15);
  int bcol = tn * 16 + (lane & 15);
  int ka = 8  * (lane >> 4);                   // A: high half-wave holds K+8 block
  int kb = 8  * (lane >> 4);                   // B: high half-wave holds K+8 block
  const __bf16* ap = A  + (size_t)arow * lda;
  const __bf16* bp = Bt + (size_t)bcol * ldb;

  v8f acc = {};
  for (int k0 = 0; k0 < Kd; k0 += 32) {
    if (k0 + 32 < Kd) {                        // warm the next K-block (L2-resident data)
      __builtin_prefetch(ap + k0 + 32 + ka, 0, 3);
      __builtin_prefetch(bp + k0 + 32 + kb, 0, 3);
    }
    v16bf a, b;
    #pragma unroll
    for (int e = 0; e < 16; ++e) a[e] = ap[k0 + (e & 7) + 16 * (e >> 3) + ka];
    #pragma unroll
    for (int e = 0; e < 16; ++e) b[e] = bp[k0 + (e & 7) + 16 * (e >> 3) + kb];
    acc = __builtin_amdgcn_wmma_f32_16x16x32_bf16(false, a, false, b, (short)0, acc, false, false);
  }

  int n = tn * 16 + (lane & 15);
  float bv = bias ? bias[n] : 0.f;
  #pragma unroll
  for (int g = 0; g < 8; ++g) {
    int m = tm * 16 + g + 8 * (lane >> 4);
    float o = acc[g] + bv;
    if (addend) o += addend[(size_t)m * ldc + n];
    C[(size_t)m * ldc + n] = o;
  }
}

// ---------------------------------------------------------------------------
// Depthwise causal conv (K=4) over sequence + SiLU. xs holds [row][256], xi = cols 0..127.
// ---------------------------------------------------------------------------
__global__ void conv_silu_kernel(const float* __restrict__ xs, const float* __restrict__ cw,
                                 const float* __restrict__ cb, float* __restrict__ xc,
                                 __bf16* __restrict__ xc_bf) {
  int idx = blockIdx.x * blockDim.x + threadIdx.x;
  if (idx >= BN * SN * EN) return;
  int e = idx % EN;
  int s = (idx / EN) % SN;
  int b = idx / (EN * SN);
  float acc = cb[e];
  const float* w = cw + e * KN;
  #pragma unroll
  for (int k = 0; k < KN; ++k) {
    int t = s - (KN - 1) + k;
    if (t >= 0) acc += w[k] * xs[((size_t)(b * SN + t)) * 256 + e];
  }
  float y = acc * sigmoidf_(acc);
  xc[idx] = y;
  xc_bf[idx] = (__bf16)y;
}

// ---------------------------------------------------------------------------
// delta = softplus(dbc[:, :R] @ W_dt + b_dt)   (R=4, tiny K -> scalar FMAs)
// ---------------------------------------------------------------------------
__global__ void delta_kernel(const float* __restrict__ dbc, const float* __restrict__ Wdt,
                             const float* __restrict__ bdt, float* __restrict__ delta) {
  int idx = blockIdx.x * blockDim.x + threadIdx.x;
  if (idx >= BN * SN * EN) return;
  int e = idx % EN;
  int row = idx / EN;
  const float* dr = dbc + (size_t)row * DBC_LD;
  float d = bdt[e];
  #pragma unroll
  for (int r = 0; r < RN; ++r) d += dr[r] * Wdt[r * EN + e];
  delta[idx] = (d > 20.f) ? d : log1pf(__expf(d));
}

// ---------------------------------------------------------------------------
// Selective scan: one wave32 per (b,e) channel; 2 of the 64 states per lane.
// h_n <- exp(delta*A_n)*h_n + (delta*x)*B_n ; y = sum_n h_n*C_n + D*x
// Serial critical path -> prefetch future timestep rows into cache while the
// recurrence FMAs retire.
// ---------------------------------------------------------------------------
__global__ void scan_kernel(const float* __restrict__ delta, const float* __restrict__ xconv,
                            const float* __restrict__ dbc, const float* __restrict__ A_log,
                            const float* __restrict__ Dv, float* __restrict__ y) {
  int lane = threadIdx.x & 31;
  int wave = (blockIdx.x * blockDim.x + threadIdx.x) >> 5;
  if (wave >= BN * EN) return;
  int b = wave / EN, e = wave % EN;
  int n0 = 2 * lane;
  float A0 = -__expf(A_log[(size_t)e * HN + n0]);
  float A1 = -__expf(A_log[(size_t)e * HN + n0 + 1]);
  float D_ = Dv[e];
  float h0 = 0.f, h1 = 0.f;
  for (int s = 0; s < SN; ++s) {
    size_t row = (size_t)b * SN + s;
    if (s + SCAN_PF < SN) {                    // wave-uniform guard, no divergence
      size_t prow = row + SCAN_PF;
      __builtin_prefetch(dbc + prow * DBC_LD + RN + n0, 0, 3);
      __builtin_prefetch(delta + prow * EN + e, 0, 3);
      __builtin_prefetch(xconv + prow * EN + e, 0, 3);
    }
    float dl = delta[row * EN + e];      // wave-uniform
    float xv = xconv[row * EN + e];      // wave-uniform
    float dx = dl * xv;
    const float* bc = dbc + row * DBC_LD;
    float Bv0 = bc[RN + n0],      Bv1 = bc[RN + n0 + 1];
    float Cv0 = bc[RN + HN + n0], Cv1 = bc[RN + HN + n0 + 1];
    h0 = __expf(dl * A0) * h0 + dx * Bv0;
    h1 = __expf(dl * A1) * h1 + dx * Bv1;
    float acc = h0 * Cv0 + h1 * Cv1;
    #pragma unroll
    for (int off = 16; off; off >>= 1) acc += __shfl_xor(acc, off, 32);
    if (lane == 0) y[row * EN + e] = acc + D_ * xv;
  }
}

// ---------------------------------------------------------------------------
// gate = y * silu(skip), converted to bf16 as A-operand of the out-proj GEMM
// ---------------------------------------------------------------------------
__global__ void gate_kernel(const float* __restrict__ xs, const float* __restrict__ ybuf,
                            __bf16* __restrict__ gate_bf) {
  int idx = blockIdx.x * blockDim.x + threadIdx.x;
  if (idx >= BN * SN * EN) return;
  int e = idx % EN;
  int row = idx / EN;
  float sk = xs[(size_t)row * 256 + 128 + e];
  gate_bf[idx] = (__bf16)(ybuf[idx] * sk * sigmoidf_(sk));
}

// ---------------------------------------------------------------------------
extern "C" void kernel_launch(void* const* d_in, const int* in_sizes, int n_in,
                              void* d_out, int out_size, void* d_ws, size_t ws_size,
                              hipStream_t stream) {
  const float* x        = (const float*)d_in[0];
  const float* norm_w   = (const float*)d_in[1];
  const float* W_in     = (const float*)d_in[2];
  const float* b_in     = (const float*)d_in[3];
  const float* conv_w   = (const float*)d_in[4];
  const float* conv_b   = (const float*)d_in[5];
  const float* W_param  = (const float*)d_in[6];
  const float* W_dt     = (const float*)d_in[7];
  const float* b_dt     = (const float*)d_in[8];
  const float* W_out    = (const float*)d_in[9];
  const float* b_out    = (const float*)d_in[10];
  const float* A_log    = (const float*)d_in[11];
  const float* Dvec     = (const float*)d_in[12];
  const float* fnw      = (const float*)d_in[13];

  // ---- workspace layout (fp32 region then bf16 region), ~8.4 MB total ----
  float* hres  = (float*)d_ws;                 // [ROWS][64]   residual stream
  float* xs    = hres  + (size_t)ROWS * HN;    // [ROWS][256]  in-proj out (xi|skip)
  float* xconv = xs    + (size_t)ROWS * 256;   // [ROWS][128]  conv+silu
  float* delta = xconv + (size_t)ROWS * EN;    // [ROWS][128]
  float* dbc   = delta + (size_t)ROWS * EN;    // [ROWS][144]  (delta_raw|B|C) padded
  float* ybuf  = dbc   + (size_t)ROWS * DBC_LD;// [ROWS][128]
  __bf16* bfb      = (__bf16*)(ybuf + (size_t)ROWS * EN);
  __bf16* xn_bf    = bfb;                        // [ROWS][64]
  __bf16* xconv_bf = xn_bf    + (size_t)ROWS*HN; // [ROWS][128]
  __bf16* gate_bf  = xconv_bf + (size_t)ROWS*EN; // [ROWS][128]
  __bf16* WinT     = gate_bf  + (size_t)ROWS*EN; // [L][256][64]
  __bf16* WparamT  = WinT     + (size_t)LN*256*HN;   // [L][144][128]
  __bf16* WoutT    = WparamT  + (size_t)LN*DBC_LD*EN;// [L][64][128]

  // ---- one-time (per launch, deterministic) weight transpose -> bf16 ----
  wconv_kernel<<<(LN*256*HN + 255)/256, 256, 0, stream>>>(W_in,    WinT,    LN, HN, 2*EN, 2*EN);
  wconv_kernel<<<(LN*DBC_LD*EN + 255)/256, 256, 0, stream>>>(W_param, WparamT, LN, EN, RN + 2*HN, DBC_LD);
  wconv_kernel<<<(LN*HN*EN + 255)/256, 256, 0, stream>>>(W_out,   WoutT,   LN, EN, HN, HN);
  copy_kernel<<<(ROWS*HN + 255)/256, 256, 0, stream>>>(x, hres, ROWS*HN);

  for (int l = 0; l < LN; ++l) {
    // 1) xn = rmsnorm(h) -> bf16
    rmsnorm_kernel<<<256, 256, 0, stream>>>(hres, norm_w + l*HN, xn_bf, nullptr);
    // 2) xs = xn @ W_in + b_in      (2048x64 @ 64x256) -> WMMA
    gemm_wmma_bf16<<<(ROWS/16)*(256/16)*32/256, 256, 0, stream>>>(
        xn_bf, HN, WinT + (size_t)l*256*HN, HN, b_in + l*256, nullptr, xs, 256, ROWS, 256, HN);
    // 3) xi = silu(causal_conv(xs[:, :128]))
    conv_silu_kernel<<<(ROWS*EN + 255)/256, 256, 0, stream>>>(
        xs, conv_w + l*EN*KN, conv_b + l*EN, xconv, xconv_bf);
    // 4) dbc = xi @ W_param          (2048x128 @ 128x132, N padded to 144) -> WMMA
    gemm_wmma_bf16<<<(ROWS/16)*(DBC_LD/16)*32/256, 256, 0, stream>>>(
        xconv_bf, EN, WparamT + (size_t)l*DBC_LD*EN, EN, nullptr, nullptr, dbc, DBC_LD, ROWS, DBC_LD, EN);
    // 5) delta = softplus(dbc[:, :4] @ W_dt + b_dt)
    delta_kernel<<<(ROWS*EN + 255)/256, 256, 0, stream>>>(dbc, W_dt + l*RN*EN, b_dt + l*EN, delta);
    // 6) selective scan (sequential over S, 256 wave32 channels)
    scan_kernel<<<(BN*EN*32)/256, 256, 0, stream>>>(
        delta, xconv, dbc, A_log + (size_t)l*EN*HN, Dvec + l*EN, ybuf);
    // 7) gate = y * silu(skip)
    gate_kernel<<<(ROWS*EN + 255)/256, 256, 0, stream>>>(xs, ybuf, gate_bf);
    // 8) h += gate @ W_out + b_out   (2048x128 @ 128x64) -> WMMA w/ fused residual
    gemm_wmma_bf16<<<(ROWS/16)*(HN/16)*32/256, 256, 0, stream>>>(
        gate_bf, EN, WoutT + (size_t)l*HN*EN, EN, b_out + l*HN, hres, hres, HN, ROWS, HN, EN);
  }
  // final rmsnorm -> fp32 output
  rmsnorm_kernel<<<256, 256, 0, stream>>>(hres, fnw, nullptr, (float*)d_out);
}
